// SelfAttention_16741782520197
// MI455X (gfx1250) — matrix operand order repaired and checked by reading, not compiled, not run on hardware
//
#include <hip/hip_runtime.h>

#define BATCH 4
#define SEQ   2048
#define DM    1024            // d_model == attn_dim
#define MTOT  (BATCH * SEQ)   // 8192 rows

typedef __attribute__((ext_vector_type(16))) __bf16        v16bf;
typedef __attribute__((ext_vector_type(8)))  float         v8f;
typedef __attribute__((ext_vector_type(4)))  unsigned int  u32x4;

__device__ __forceinline__ unsigned short f2bf(float f) {
  unsigned int u = __float_as_uint(f);
  u += 0x7FFFu + ((u >> 16) & 1u);           // round-to-nearest-even
  return (unsigned short)(u >> 16);
}

// Build a 16-element bf16 fragment from two 16-byte chunks.
__device__ __forceinline__ v16bf load_frag(const unsigned short* p0,
                                           const unsigned short* p1) {
  union { u32x4 q[2]; v16bf v; } f;
  f.q[0] = *(const u32x4*)p0;
  f.q[1] = *(const u32x4*)p1;
  return f.v;
}

// ---------------- conversion pre-passes ----------------

__global__ void cvt_f32_bf16(const float* __restrict__ in,
                             unsigned short* __restrict__ out, int n) {
  int i = blockIdx.x * blockDim.x + threadIdx.x;
  if (i < n) out[i] = f2bf(in[i]);
}

// W[k][n] fp32  ->  Wt[n][k] bf16   (DM x DM)
__global__ void cvt_w_T(const float* __restrict__ W,
                        unsigned short* __restrict__ Wt) {
  int i = blockIdx.x * blockDim.x + threadIdx.x;  // i = n*DM + k
  int n = i >> 10, k = i & 1023;
  Wt[i] = f2bf(W[k * DM + n]);
}

// ---------------- QKV projection GEMM (bf16 WMMA, fp32 acc) ----------------
// C[m][n] = sum_k X[m][k] * W[k][n] + bias[n]
// block = 256 threads (8 waves); wave -> 16x64 tile; grid = (MTOT/16, DM/512)
// Software-pipelined: next k-chunk's fragments are prefetched while the
// current chunk's 4 WMMAs execute, so s_wait_loadcnt can be partial.

__global__ void __launch_bounds__(256)
proj_gemm(const unsigned short* __restrict__ Xb,   // [MTOT][DM] bf16
          const unsigned short* __restrict__ Wt,   // [DM(n)][DM(k)] bf16
          const float* __restrict__ bias,          // [DM]
          unsigned short* __restrict__ Out,        // [MTOT][DM] bf16 (or null)
          unsigned short* __restrict__ OutT) {     // [BATCH][DM][SEQ] bf16 (or null)
  const int lane  = threadIdx.x & 31;
  const int wave  = threadIdx.x >> 5;
  const int r15   = lane & 15;
  const int g     = lane >> 4;
  const int mbase = blockIdx.x * 16;
  const int nbase = blockIdx.y * 512 + wave * 64;

  v8f acc[4] = {};

  const unsigned short* arow = Xb + (size_t)(mbase + r15) * DM + g * 8;
  const unsigned short* brow = Wt + (size_t)(nbase + r15) * DM + g * 8;

  // prologue: fragments for kc = 0
  v16bf a_cur = load_frag(arow, arow + 16);
  v16bf b_cur[4];
  #pragma unroll
  for (int t = 0; t < 4; ++t) {
    const unsigned short* pb = brow + (size_t)t * 16 * DM;
    b_cur[t] = load_frag(pb, pb + 16);
  }

  #pragma unroll 2
  for (int kc = 0; kc < DM; kc += 32) {
    const int kn = (kc + 32 < DM) ? kc + 32 : kc;  // last iter: dummy reload
    const unsigned short* pa = arow + kn;
    v16bf a_nxt = load_frag(pa, pa + 16);
    v16bf b_nxt[4];
    #pragma unroll
    for (int t = 0; t < 4; ++t) {
      const unsigned short* pb = brow + (size_t)t * 16 * DM + kn;
      b_nxt[t] = load_frag(pb, pb + 16);
    }
    #pragma unroll
    for (int t = 0; t < 4; ++t)
      acc[t] = __builtin_amdgcn_wmma_f32_16x16x32_bf16(
          false, a_cur, false, b_cur[t], (short)0, acc[t], false, false);
    a_cur = a_nxt;
    #pragma unroll
    for (int t = 0; t < 4; ++t) b_cur[t] = b_nxt[t];
  }

  #pragma unroll
  for (int t = 0; t < 4; ++t) {
    int n = nbase + t * 16 + r15;
    float bv = bias[n];
    #pragma unroll
    for (int r = 0; r < 8; ++r) {
      int m = mbase + r + g * 8;
      unsigned short h = f2bf(acc[t][r] + bv);
      if (Out)  Out[(size_t)m * DM + n] = h;
      if (OutT) {
        int b = m >> 11;            // m / SEQ
        int s = m & (SEQ - 1);      // m % SEQ
        OutT[((size_t)b * DM + n) * SEQ + s] = h;
      }
    }
  }
}

// ---------------- causal flash attention ----------------
// block = 256 threads (8 waves); one 16-row query tile per block.
// Key tiles of 128: wave w computes score subtile for keys [kbase+16w, +16),
// softmax is done cooperatively through LDS, then each wave does PV for its
// private 128-column output slice. Score and PV loops are software-pipelined.

__global__ void __launch_bounds__(256)
attn_fwd(const unsigned short* __restrict__ Qb,   // [B][S][DM] bf16
         const unsigned short* __restrict__ Kb,   // [B][S][DM] bf16
         const unsigned short* __restrict__ Vt,   // [B][DM][S] bf16
         float* __restrict__ Out) {               // [B][S][DM] f32
  const int lane  = threadIdx.x & 31;
  const int wave  = threadIdx.x >> 5;
  const int r15   = lane & 15;
  const int g     = lane >> 4;
  const int qbase = blockIdx.x * 16;
  const int batch = blockIdx.y;
  const float inv_scale = 0.03125f;               // 1/sqrt(1024)

  __shared__ float          sS[16 * 128];
  __shared__ unsigned short sP[16 * 128];
  __shared__ float sM[16], sL[16], sScale[16];

  if (threadIdx.x < 16) { sM[threadIdx.x] = -__builtin_inff(); sL[threadIdx.x] = 0.f; }

  v8f oacc[8] = {};                  // 16 x 128 slice, cols wave*128 + t*16
  const int wcol = wave * 128;

  const unsigned short* Qrow =
      Qb + ((size_t)batch * SEQ + qbase + r15) * DM + g * 8;

  const int nkt = (qbase + 15) / 128 + 1;          // causal: tiles up to diagonal
  for (int kt = 0; kt < nkt; ++kt) {
    const int kbase = kt * 128;

    // ---- scores: this wave handles keys [kbase + wave*16, +16)
    v8f sacc = {};
    const unsigned short* Krow =
        Kb + ((size_t)batch * SEQ + kbase + wave * 16 + r15) * DM + g * 8;
    v16bf a_cur = load_frag(Qrow, Qrow + 16);
    v16bf b_cur = load_frag(Krow, Krow + 16);
    #pragma unroll 4
    for (int kc = 0; kc < DM; kc += 32) {
      const int kn = (kc + 32 < DM) ? kc + 32 : kc;
      v16bf a_nxt = load_frag(Qrow + kn, Qrow + kn + 16);
      v16bf b_nxt = load_frag(Krow + kn, Krow + kn + 16);
      sacc = __builtin_amdgcn_wmma_f32_16x16x32_bf16(
          false, a_cur, false, b_cur, (short)0, sacc, false, false);
      a_cur = a_nxt;
      b_cur = b_nxt;
    }
    #pragma unroll
    for (int r = 0; r < 8; ++r)
      sS[(r + g * 8) * 128 + wave * 16 + r15] = sacc[r] * inv_scale;
    __syncthreads();

    // ---- online softmax: 16 threads (one half-wave) per row
    {
      const int r    = threadIdx.x >> 4;   // 0..15
      const int c0   = threadIdx.x & 15;
      const int grow = qbase + r;
      float v[8];
      float mx = -__builtin_inff();
      #pragma unroll
      for (int i = 0; i < 8; ++i) {
        int col = c0 + i * 16;
        float x = sS[r * 128 + col];
        if (kbase + col > grow) x = -__builtin_inff();   // causal mask
        v[i] = x;
        mx = fmaxf(mx, x);
      }
      #pragma unroll
      for (int off = 8; off >= 1; off >>= 1)
        mx = fmaxf(mx, __shfl_xor(mx, off, 16));
      float mprev = sM[r];
      float mnew  = fmaxf(mprev, mx);
      float scale = __expf(mprev - mnew);
      float sum = 0.f;
      #pragma unroll
      for (int i = 0; i < 8; ++i) {
        int col = c0 + i * 16;
        float p = __expf(v[i] - mnew);
        sP[r * 128 + col] = f2bf(p);
        sum += p;
      }
      #pragma unroll
      for (int off = 8; off >= 1; off >>= 1)
        sum += __shfl_xor(sum, off, 16);
      if (c0 == 0) {
        sM[r]     = mnew;
        sL[r]     = sL[r] * scale + sum;
        sScale[r] = scale;
      }
    }
    __syncthreads();

    // ---- rescale accumulators
    #pragma unroll
    for (int r = 0; r < 8; ++r) {
      float sc = sScale[r + g * 8];
      #pragma unroll
      for (int t = 0; t < 8; ++t) oacc[t][r] *= sc;
    }

    // ---- PV: flattened (pc,t) pipeline; P A-frags from LDS, V B-frags
    //      double-buffered from global (V^T layout -> two contiguous 16B chunks).
    {
      v16bf pA[4];
      #pragma unroll
      for (int pc = 0; pc < 4; ++pc) {
        const unsigned short* pp = &sP[r15 * 128 + pc * 32 + g * 8];
        pA[pc] = load_frag(pp, pp + 16);
      }
      const unsigned short* Vbase =
          Vt + ((size_t)batch * DM + wcol + r15) * SEQ + kbase + g * 8;
      // i = pc*8 + t ; fragment address = Vbase + t*16*SEQ + pc*32
      v16bf vB_cur = load_frag(Vbase, Vbase + 16);
      #pragma unroll
      for (int i = 0; i < 32; ++i) {
        const int inx = (i + 1 < 32) ? i + 1 : i;
        const unsigned short* pv =
            Vbase + (size_t)(inx & 7) * 16 * SEQ + (inx >> 3) * 32;
        v16bf vB_nxt = load_frag(pv, pv + 16);
        oacc[i & 7] = __builtin_amdgcn_wmma_f32_16x16x32_bf16(
            false, pA[i >> 3], false, vB_cur, (short)0, oacc[i & 7], false, false);
        vB_cur = vB_nxt;
      }
    }
    __syncthreads();
  }

  // ---- finalize: divide by row sums, write fp32 output
  #pragma unroll
  for (int r = 0; r < 8; ++r) {
    int   row = r + g * 8;
    float inv = 1.0f / sL[row];
    int  grow = qbase + row;
    #pragma unroll
    for (int t = 0; t < 8; ++t)
      Out[((size_t)batch * SEQ + grow) * DM + wcol + t * 16 + r15] =
          oacc[t][r] * inv;
  }
}

// ---------------- host launcher ----------------

extern "C" void kernel_launch(void* const* d_in, const int* in_sizes, int n_in,
                              void* d_out, int out_size, void* d_ws, size_t ws_size,
                              hipStream_t stream) {
  (void)in_sizes; (void)n_in; (void)out_size; (void)ws_size;

  const float* X  = (const float*)d_in[0];
  const float* Wq = (const float*)d_in[1];
  const float* bq = (const float*)d_in[2];
  const float* Wk = (const float*)d_in[3];
  const float* bk = (const float*)d_in[4];
  const float* Wv = (const float*)d_in[5];
  const float* bv = (const float*)d_in[6];
  float* out = (float*)d_out;

  // workspace layout (bf16 halves): Xb | Wtq | Wtk | Wtv | Qb | Kb | Vt
  unsigned short* Xb  = (unsigned short*)d_ws;
  unsigned short* Wtq = Xb  + (size_t)MTOT * DM;
  unsigned short* Wtk = Wtq + (size_t)DM * DM;
  unsigned short* Wtv = Wtk + (size_t)DM * DM;
  unsigned short* Qb  = Wtv + (size_t)DM * DM;
  unsigned short* Kb  = Qb  + (size_t)MTOT * DM;
  unsigned short* Vt  = Kb  + (size_t)MTOT * DM;

  cvt_f32_bf16<<<(MTOT * DM) / 256, 256, 0, stream>>>(X, Xb, MTOT * DM);
  cvt_w_T<<<(DM * DM) / 256, 256, 0, stream>>>(Wq, Wtq);
  cvt_w_T<<<(DM * DM) / 256, 256, 0, stream>>>(Wk, Wtk);
  cvt_w_T<<<(DM * DM) / 256, 256, 0, stream>>>(Wv, Wtv);

  dim3 gg(MTOT / 16, DM / 512);
  proj_gemm<<<gg, 256, 0, stream>>>(Xb, Wtq, bq, Qb, nullptr);
  proj_gemm<<<gg, 256, 0, stream>>>(Xb, Wtk, bk, Kb, nullptr);
  proj_gemm<<<gg, 256, 0, stream>>>(Xb, Wtv, bv, nullptr, Vt);

  dim3 ga(SEQ / 16, BATCH);
  attn_fwd<<<ga, 256, 0, stream>>>(Qb, Kb, Vt, out);
}
